// Dlrm_83794811944966
// MI455X (gfx1250) — matrix operand order repaired and checked by reading, not compiled, not run on hardware
//
#include <hip/hip_runtime.h>
#include <hip/hip_bf16.h>
#include <math.h>

// ---------------------------------------------------------------------------
// DLRM forward for MI455X (gfx1250, wave32).
// f16 data / f32 accumulate via v_wmma_f32_16x16x32_f16.
// ---------------------------------------------------------------------------

typedef __attribute__((ext_vector_type(16))) _Float16 v16h;
typedef __attribute__((ext_vector_type(8)))  _Float16 v8h;
typedef __attribute__((ext_vector_type(8)))  float    v8f;

#define B_ROWS   32768
#define EMB_D    128
#define N_TABLES 26
#define VOCAB    50000

// Build a 16-bit A/B fragment from two 16-byte LDS chunks.
static __device__ __forceinline__ v16h mk_frag(const _Float16* plo, const _Float16* phi) {
  v8h lo = *(const v8h*)plo;
  v8h hi = *(const v8h*)phi;
  v16h r;
#pragma unroll
  for (int i = 0; i < 8; ++i) { r[i] = lo[i]; r[i + 8] = hi[i]; }
  return r;
}

static __device__ __forceinline__ v8f wmma_f16(v16h a, v16h b, v8f c) {
  // D = A(16x32 f16) * B(32x16 f16) + C(16x16 f32)
  return __builtin_amdgcn_wmma_f32_16x16x32_f16(false, a, false, b, (short)0, c,
                                                false, false);
}

// ---------------------------------------------------------------------------
// Weight convert: f32 row-major (K x N) -> f16 transposed (N x Kpad), zero-pad.
// Transposed storage makes the WMMA B fragment two contiguous 16B reads per lane.
// ---------------------------------------------------------------------------
__global__ void k_convert_wT(const float* __restrict__ W, _Float16* __restrict__ Wt,
                             int K, int N, int Kpad) {
  const int i = blockIdx.x * blockDim.x + threadIdx.x;
  if (i >= N * Kpad) return;
  const int n = i / Kpad;
  const int k = i - n * Kpad;
  Wt[i] = (k < K) ? (_Float16)W[(size_t)k * N + n] : (_Float16)0.f;
}

// dense_x (32768 x 13 f32) -> f16, padded to width 32 (cols 13..31 = 0)
__global__ void k_pack_dense(const float* __restrict__ x, _Float16* __restrict__ a0) {
  const int i = blockIdx.x * blockDim.x + threadIdx.x;  // 32768*32 threads
  const int r = i >> 5, c = i & 31;
  a0[i] = (c < 13) ? (_Float16)x[r * 13 + c] : (_Float16)0.f;
}

// ---------------------------------------------------------------------------
// Fused GEMM + bias + ReLU:  C[M x N](f16) = relu(A[M x K](f16) @ W + bias)
// W supplied transposed f16 (N x K). Block: 256 threads = 8 waves.
// Block tile 128(M) x 128(N); wave (wm,wn) in 4x2 grid owns a 32x64 sub-tile:
//   2 A-frags x 4 B-frags -> 8 back-to-back WMMAs per 32-wide k-step.
// All 12 fragment ds_loads issue first, then the 8 WMMAs (single dscnt wait).
// LDS double buffered; row stride 40 halves -> conflict-free b128 frag reads.
// ---------------------------------------------------------------------------
#define GMT 128
#define GNT 128
#define LSTR 40

__global__ __launch_bounds__(256) void k_gemm_relu(
    const _Float16* __restrict__ A, const _Float16* __restrict__ Wt,
    const float* __restrict__ bias, _Float16* __restrict__ C,
    int N, int K) {
  __shared__ _Float16 As[2][GMT][LSTR];
  __shared__ _Float16 Ws[2][GNT][LSTR];

  const int tid  = threadIdx.x;
  const int w    = tid >> 5;
  const int wm   = w & 3;       // M strip: 32 rows
  const int wn   = w >> 2;      // N strip: 64 cols
  const int lane = tid & 31;
  const int kh   = lane >> 4;   // K-half select (WMMA lane layout)
  const int lm   = lane & 15;   // M (A frag) / N (B frag) within 16x16 tile
  const int m0   = blockIdx.x * GMT;
  const int n0   = blockIdx.y * GNT;
  const int nk   = K >> 5;      // K is always a multiple of 32 (pre-padded)

  v8f acc[2][4];
#pragma unroll
  for (int mt = 0; mt < 2; ++mt)
#pragma unroll
    for (int nt = 0; nt < 4; ++nt)
#pragma unroll
      for (int r = 0; r < 8; ++r) acc[mt][nt][r] = 0.f;

  // ---- prologue: stage 0 (A tile: 512 uint4 items, W tile: 512 items) ----
  {
#pragma unroll
    for (int it = 0; it < 2; ++it) {
      const int item = tid + it * 256;
      const int row = item >> 2, c8 = (item & 3) << 3;
      *(uint4*)&As[0][row][c8] = *(const uint4*)(A + (size_t)(m0 + row) * K + c8);
      *(uint4*)&Ws[0][row][c8] = *(const uint4*)(Wt + (size_t)(n0 + row) * K + c8);
    }
  }
  __syncthreads();

  for (int ks = 0; ks < nk; ++ks) {
    const int buf = ks & 1;
    const bool has_next = (ks + 1) < nk;

    uint4 ra[2], rw[2];
    ra[0] = make_uint4(0, 0, 0, 0); ra[1] = ra[0]; rw[0] = ra[0]; rw[1] = ra[0];
    if (has_next) {
      const int k0 = (ks + 1) << 5;
#pragma unroll
      for (int it = 0; it < 2; ++it) {
        const int item = tid + it * 256;
        const int row = item >> 2, c8 = (item & 3) << 3;
        ra[it] = *(const uint4*)(A + (size_t)(m0 + row) * K + k0 + c8);
        rw[it] = *(const uint4*)(Wt + (size_t)(n0 + row) * K + k0 + c8);
      }
      if (ks + 2 < nk)  // warm L2/L0 for the tile after next (global_prefetch_b8)
        __builtin_prefetch(Wt + (size_t)(n0 + (tid >> 2)) * K + ((ks + 2) << 5), 0, 1);
    }

    // ---- load ALL fragments first (12 ds_load_b128), then 8 WMMAs ----
    v16h af[2];
#pragma unroll
    for (int mt = 0; mt < 2; ++mt) {
      const _Float16* ab = &As[buf][(wm << 5) + (mt << 4) + lm][0];
      af[mt] = mk_frag(ab + (kh << 3), ab + 16 + (kh << 3));
    }
    v16h bf[4];
#pragma unroll
    for (int nt = 0; nt < 4; ++nt) {
      const _Float16* bb = &Ws[buf][(wn << 6) + (nt << 4) + lm][kh << 3];
      bf[nt] = mk_frag(bb, bb + 16);
    }
#pragma unroll
    for (int mt = 0; mt < 2; ++mt)
#pragma unroll
      for (int nt = 0; nt < 4; ++nt)
        acc[mt][nt] = wmma_f16(af[mt], bf[nt], acc[mt][nt]);

    // ---- commit next stage ----
    if (has_next) {
      const int nb = buf ^ 1;
#pragma unroll
      for (int it = 0; it < 2; ++it) {
        const int item = tid + it * 256;
        const int row = item >> 2, c8 = (item & 3) << 3;
        *(uint4*)&As[nb][row][c8] = ra[it];
        *(uint4*)&Ws[nb][row][c8] = rw[it];
      }
    }
    __syncthreads();
  }

  // ---- epilogue: f32 bias + ReLU, store f16 ----
#pragma unroll
  for (int mt = 0; mt < 2; ++mt) {
#pragma unroll
    for (int nt = 0; nt < 4; ++nt) {
      const int col = n0 + (wn << 6) + (nt << 4) + lm;
      const float bv = bias[col];
#pragma unroll
      for (int r = 0; r < 8; ++r) {
        const int row = m0 + (wm << 5) + (mt << 4) + (kh << 3) + r;
        const float v = acc[mt][nt][r] + bv;
        C[(size_t)row * N + col] = (_Float16)(v > 0.f ? v : 0.f);
      }
    }
  }
}

// ---------------------------------------------------------------------------
// Embedding gather + pairwise interaction.
// One sample per wave (4 samples / 128-thread block).
// F = [bot_out; emb rows 1..26; zero pad] : 32 x 128 f16 in LDS.
// Z = F*F^T via 3 lower-triangle 16x16 WMMA tiles, K=128 (4 steps).
// Emits xcat row: [bot(128) | 351 strict-lower-tri | zero pad to 512].
// ---------------------------------------------------------------------------
#define IW   4
#define FSTR 136   // row stride (halves): 272B, bank-conflict free B-frag reads
#define ZSTR 36

__global__ __launch_bounds__(128) void k_interact(
    const _Float16* __restrict__ bot, const int* __restrict__ sparse,
    const float* __restrict__ emb, _Float16* __restrict__ xcat) {
  __shared__ _Float16 F[IW][32][FSTR];
  __shared__ float    Z[IW][32][ZSTR];

  const int tid = threadIdx.x, w = tid >> 5, lane = tid & 31;
  const int kh = lane >> 4, lm = lane & 15;
  const int s = blockIdx.x * IW + w;

  // row 0: bottom-MLP output (already f16, contiguous 8B per lane)
  *(uint2*)&F[w][0][lane << 2] = *(const uint2*)(bot + (size_t)s * 128 + (lane << 2));

  // rows 1..26: gather (f32 -> f16); this is the HBM-dominant stream (~436 MB)
#pragma unroll 1
  for (int t = 0; t < N_TABLES; ++t) {
    const int idx = sparse[s * N_TABLES + t];
    const float4 v =
        *(const float4*)(emb + ((size_t)t * VOCAB + (size_t)idx) * EMB_D + (lane << 2));
    _Float16* d = &F[w][t + 1][lane << 2];
    d[0] = (_Float16)v.x; d[1] = (_Float16)v.y; d[2] = (_Float16)v.z; d[3] = (_Float16)v.w;
  }

  // rows 27..31: zero pad (so 32x32 WMMA tiles are valid)
#pragma unroll
  for (int r = 27; r < 32; ++r) {
    uint2 z2; z2.x = 0u; z2.y = 0u;
    *(uint2*)&F[w][r][lane << 2] = z2;
  }

  // Z = F * F^T : tiles (mt,nt) = (0,0), (1,0), (1,1)
  v8f zA, zB, zC;
#pragma unroll
  for (int r = 0; r < 8; ++r) { zA[r] = 0.f; zB[r] = 0.f; zC[r] = 0.f; }

#pragma unroll
  for (int ks = 0; ks < 4; ++ks) {
    const int kb = ks << 5;
    const _Float16* a0 = &F[w][lm][kb];
    const _Float16* a1 = &F[w][16 + lm][kb];
    const v16h af0 = mk_frag(a0 + (kh << 3), a0 + 16 + (kh << 3));
    const v16h af1 = mk_frag(a1 + (kh << 3), a1 + 16 + (kh << 3));
    // B = F^T, so the B fragment is two 8-element chunks of row lm of F (no transpose)
    const _Float16* b0 = &F[w][lm][kb + (kh << 3)];
    const _Float16* b1 = &F[w][16 + lm][kb + (kh << 3)];
    const v16h bf0 = mk_frag(b0, b0 + 16);
    const v16h bf1 = mk_frag(b1, b1 + 16);
    zA = wmma_f16(af0, bf0, zA);
    zB = wmma_f16(af1, bf0, zB);
    zC = wmma_f16(af1, bf1, zC);
  }

  // spill Z tiles to LDS for the irregular triangle extraction
#pragma unroll
  for (int r = 0; r < 8; ++r) {
    Z[w][(kh << 3) + r][lm]           = zA[r];
    Z[w][16 + (kh << 3) + r][lm]      = zB[r];
    Z[w][16 + (kh << 3) + r][16 + lm] = zC[r];
  }

  _Float16* xrow = xcat + (size_t)s * 512;

  // strict lower triangle, 351 pairs (i in 1..26, j < i), row-of-pairs order
  for (int p = lane; p < 351; p += 32) {
    int i = (int)((1.0f + sqrtf(1.0f + 8.0f * (float)p)) * 0.5f);
    while (i * (i - 1) / 2 > p) --i;
    while ((i + 1) * i / 2 <= p) ++i;
    const int j = p - i * (i - 1) / 2;
    xrow[128 + p] = (_Float16)Z[w][i][j];
  }

  // cols 0..127 = bottom output; cols 479..511 = 0 (matches zero-padded tw0)
  *(uint2*)(xrow + (lane << 2)) = *(const uint2*)&F[w][0][lane << 2];
  for (int c = 479 + lane; c < 512; c += 32) xrow[c] = (_Float16)0.f;
}

// ---------------------------------------------------------------------------
// Final layer: out[r] = relu(t4[r,:] . tw4 + tb4), N=1 -> plain dot per thread.
// ---------------------------------------------------------------------------
__global__ __launch_bounds__(256) void k_final(
    const _Float16* __restrict__ t4, const float* __restrict__ w4,
    const float* __restrict__ b4, float* __restrict__ out) {
  const int r = blockIdx.x * 256 + threadIdx.x;
  const _Float16* x = t4 + (size_t)r * 256;
  float acc = b4[0];
#pragma unroll 8
  for (int k = 0; k < 256; ++k) acc += (float)x[k] * w4[k];
  out[r] = acc > 0.f ? acc : 0.f;
}

// ---------------------------------------------------------------------------
extern "C" void kernel_launch(void* const* d_in, const int* in_sizes, int n_in,
                              void* d_out, int out_size, void* d_ws, size_t ws_size,
                              hipStream_t stream) {
  const float* dense  = (const float*)d_in[0];
  const int*   sparse = (const int*)d_in[1];
  const float* emb    = (const float*)d_in[2];
  const float* bw0 = (const float*)d_in[3];  const float* bb0 = (const float*)d_in[4];
  const float* bw1 = (const float*)d_in[5];  const float* bb1 = (const float*)d_in[6];
  const float* bw2 = (const float*)d_in[7];  const float* bb2 = (const float*)d_in[8];
  const float* tw0 = (const float*)d_in[9];  const float* tb0 = (const float*)d_in[10];
  const float* tw1 = (const float*)d_in[11]; const float* tb1 = (const float*)d_in[12];
  const float* tw2 = (const float*)d_in[13]; const float* tb2 = (const float*)d_in[14];
  const float* tw3 = (const float*)d_in[15]; const float* tb3 = (const float*)d_in[16];
  const float* tw4 = (const float*)d_in[17]; const float* tb4 = (const float*)d_in[18];
  float* out = (float*)d_out;

  // Workspace layout (f16 elements). Weights transposed+padded, then two
  // ping-pong activation buffers (32768 x 1024 each). Total ~135 MB.
  _Float16* base = (_Float16*)d_ws;
  _Float16* wB0T = base;                        // 512  x 32
  _Float16* wB1T = wB0T + 512 * 32;             // 256  x 512
  _Float16* wB2T = wB1T + 256 * 512;            // 128  x 256
  _Float16* wT0T = wB2T + 128 * 256;            // 1024 x 512
  _Float16* wT1T = wT0T + 1024 * 512;           // 1024 x 1024
  _Float16* wT2T = wT1T + 1024 * 1024;          // 512  x 1024
  _Float16* wT3T = wT2T + 512 * 1024;           // 256  x 512
  _Float16* act0 = wT3T + 256 * 512;            // 32768 x 32
  _Float16* bufA = act0 + (size_t)B_ROWS * 32;  // 32768 x 1024
  _Float16* bufB = bufA + (size_t)B_ROWS * 1024;

  auto cvt = [&](const float* W, _Float16* Wt, int K, int N, int Kpad) {
    const int total = N * Kpad;
    k_convert_wT<<<(total + 255) / 256, 256, 0, stream>>>(W, Wt, K, N, Kpad);
  };
  cvt(bw0, wB0T, 13,   512,  32);
  cvt(bw1, wB1T, 512,  256,  512);
  cvt(bw2, wB2T, 256,  128,  256);
  cvt(tw0, wT0T, 479,  1024, 512);
  cvt(tw1, wT1T, 1024, 1024, 1024);
  cvt(tw2, wT2T, 1024, 512,  1024);
  cvt(tw3, wT3T, 512,  256,  512);
  k_pack_dense<<<(B_ROWS * 32) / 256, 256, 0, stream>>>(dense, act0);

  auto gemm = [&](const _Float16* A, const _Float16* Wt, const float* bias,
                  _Float16* C, int N, int K) {
    dim3 grid(B_ROWS / GMT, N / GNT);
    k_gemm_relu<<<grid, 256, 0, stream>>>(A, Wt, bias, C, N, K);
  };

  // bottom MLP: 32 -> 512 -> 256 -> 128
  gemm(act0, wB0T, bb0, bufB, 512, 32);
  gemm(bufB, wB1T, bb1, bufA, 256, 512);
  gemm(bufA, wB2T, bb2, bufB, 128, 256);
  // gather + interaction -> xcat (32768 x 512, padded)
  k_interact<<<B_ROWS / IW, 128, 0, stream>>>(bufB, sparse, emb, bufA);
  // top MLP: 512 -> 1024 -> 1024 -> 512 -> 256
  gemm(bufA, wT0T, tb0, bufB, 1024, 512);
  gemm(bufB, wT1T, tb1, bufA, 1024, 1024);
  gemm(bufA, wT2T, tb2, bufB, 512, 1024);
  gemm(bufB, wT3T, tb3, bufA, 256, 512);
  // 256 -> 1 + relu
  k_final<<<B_ROWS / 256, 256, 0, stream>>>(bufA, tw4, tb4, out);

  (void)in_sizes; (void)n_in; (void)out_size; (void)ws_size;
}
